// FLASH_ShareA_FFConvM_FlashAttn_36318243455267
// MI455X (gfx1250) — matrix operand order repaired and checked by reading, not compiled, not run on hardware
//
#include <hip/hip_runtime.h>
#include <cstdint>
#include <cstddef>

// ---------------------------------------------------------------------------
// Types for CDNA5 WMMA (gfx1250, wave32)
// ---------------------------------------------------------------------------
typedef __bf16 bf16_t;
typedef __attribute__((ext_vector_type(16))) __bf16 v16bf;
typedef __attribute__((ext_vector_type(8)))  float  v8f;

struct __align__(16) U128 { unsigned int w[4]; };
union FragBF { v16bf v; U128 q[2]; };

__device__ __forceinline__ bf16_t f2bf(float f) {
  unsigned int u = __builtin_bit_cast(unsigned int, f);
  unsigned int r = u + 0x7FFFu + ((u >> 16) & 1u);   // round-to-nearest-even
  unsigned short s = (unsigned short)(r >> 16);
  return __builtin_bit_cast(bf16_t, s);
}
__device__ __forceinline__ float bf2f(bf16_t b) {
  unsigned short s = __builtin_bit_cast(unsigned short, b);
  unsigned int u = ((unsigned int)s) << 16;
  return __builtin_bit_cast(float, u);
}
__device__ __forceinline__ float silu_f(float x) { return x / (1.f + __expf(-x)); }
__device__ __forceinline__ float sigm_f(float x) { return 1.f / (1.f + __expf(-x)); }

// ---------------------------------------------------------------------------
// Synchronous LDS staging helpers (used where a transpose is required).
// LDS tiles are always 128 (frag-index) x 32 (k), row stride 32 bf16.
// ---------------------------------------------------------------------------
__device__ __forceinline__ void stage_rm(bf16_t* dst, const bf16_t* src,
                                         long srcStride, int tid) {
  int r  = tid >> 1;
  int kc = (tid & 1) * 16;
  const U128* s = reinterpret_cast<const U128*>(src + (long)r * srcStride + kc);
  U128* d = reinterpret_cast<U128*>(dst + r * 32 + kc);
  d[0] = s[0];
  d[1] = s[1];
}
// src tile is [k(32 rows)][col(128)] with row stride srcStride; LDS gets [col][k].
__device__ __forceinline__ void stage_tr(bf16_t* dst, const bf16_t* src,
                                         long srcStride, int tid) {
  int r  = tid >> 3;            // k row 0..31
  int c0 = (tid & 7) * 16;      // column chunk
  const bf16_t* s = src + (long)r * srcStride + c0;
  bf16_t tmp[16];
  *reinterpret_cast<U128*>(tmp)     = *reinterpret_cast<const U128*>(s);
  *reinterpret_cast<U128*>(tmp + 8) = *reinterpret_cast<const U128*>(s + 8);
#pragma unroll
  for (int i = 0; i < 16; ++i) dst[(c0 + i) * 32 + r] = tmp[i];
}

// ---------------------------------------------------------------------------
// Asynchronous global->LDS staging (CDNA5 GLOBAL_LOAD_ASYNC_TO_LDS_B128,
// tracked by ASYNCcnt; no VGPR round trip).  Each thread moves its own
// 32-byte chunk of the 128x32 bf16 tile with two async b128 transfers.
// The instruction offset is added to BOTH the LDS and the global address,
// so "offset:16" covers the second 16-byte chunk on both sides.
// Flat pointers to __shared__ truncate to the raw LDS byte address
// ---------------------------------------------------------------------------
__device__ __forceinline__ void stage_rm_async(bf16_t* dst, const bf16_t* src,
                                               long srcStride, int tid) {
  int r  = tid >> 1;
  int kc = (tid & 1) * 16;
  unsigned long long g =
      (unsigned long long)(uintptr_t)(src + (long)r * srcStride + kc);
  unsigned lds = (unsigned)(uintptr_t)(dst + r * 32 + kc);
  asm volatile("global_load_async_to_lds_b128 %0, %1, off"
               :: "v"(lds), "v"(g) : "memory");
  asm volatile("global_load_async_to_lds_b128 %0, %1, off offset:16"
               :: "v"(lds), "v"(g) : "memory");
}
__device__ __forceinline__ void wait_async4() {
  asm volatile("s_wait_asynccnt 0x4" ::: "memory");
}
__device__ __forceinline__ void wait_async0() {
  asm volatile("s_wait_asynccnt 0x0" ::: "memory");
}

// ---------------------------------------------------------------------------
// WMMA fragment loads per CDNA5 ISA VGPR layouts (16-bit A 16x32, B 32x16).
// A: lane m = lane&15; khalf = lane>>4; k chunks [8h,8h+8) and [16+8h,16+8h+8).
// B: lane n = lane&15; khalf = lane>>4; k chunks [8h,8h+8) and [16+8h,16+8h+8).
// ---------------------------------------------------------------------------
__device__ __forceinline__ FragBF loadFragA(const bf16_t* sA, int mrow, int lane) {
  FragBF f;
  int r = lane & 15, kh = lane >> 4;
  const bf16_t* base = sA + (mrow + r) * 32;
  f.q[0] = *reinterpret_cast<const U128*>(base + kh * 8);
  f.q[1] = *reinterpret_cast<const U128*>(base + 16 + kh * 8);
  return f;
}
__device__ __forceinline__ FragBF loadFragB(const bf16_t* sB, int ncol, int lane) {
  FragBF f;
  int r = lane & 15, kh = lane >> 4;
  const bf16_t* base = sB + (ncol + r) * 32;
  f.q[0] = *reinterpret_cast<const U128*>(base + kh * 8);
  f.q[1] = *reinterpret_cast<const U128*>(base + 16 + kh * 8);
  return f;
}

// One k-step of a 128x128 block tile: this wave's 32x64 subtile (2x4 tiles).
__device__ __forceinline__ void mma_step(const bf16_t* sA, const bf16_t* sB,
                                         int wm, int wn, int lane, v8f acc[2][4]) {
  FragBF a0 = loadFragA(sA, wm,      lane);
  FragBF a1 = loadFragA(sA, wm + 16, lane);
  FragBF b[4];
#pragma unroll
  for (int j = 0; j < 4; ++j) b[j] = loadFragB(sB, wn + j * 16, lane);
#pragma unroll
  for (int j = 0; j < 4; ++j) {
    acc[0][j] = __builtin_amdgcn_wmma_f32_16x16x32_bf16(
        false, a0.v, false, b[j].v, (short)0, acc[0][j], false, false);
    acc[1][j] = __builtin_amdgcn_wmma_f32_16x16x32_bf16(
        false, a1.v, false, b[j].v, (short)0, acc[1][j], false, false);
  }
}

__device__ __forceinline__ float block_reduce(float v, float* red, int tid, int T) {
  red[tid] = v;
  __syncthreads();
  for (int s = T >> 1; s > 0; s >>= 1) {
    if (tid < s) red[tid] += red[tid + s];
    __syncthreads();
  }
  float r = red[0];
  __syncthreads();
  return r;
}

// ---------------------------------------------------------------------------
// Kernel 1: token shift + LayerNorm statistics (affine folded into weights).
// ---------------------------------------------------------------------------
__global__ void k_norm_shift(const float* __restrict__ x, bf16_t* __restrict__ out,
                             int N, int D) {
  int row = blockIdx.x;
  int b = row / N, n = row % N;
  int tid = threadIdx.x;
  __shared__ float red[128];
  const int half = D >> 1;
  float vals[4];
  float sum = 0.f;
#pragma unroll
  for (int i = 0; i < 4; ++i) {
    int d = tid + i * 128;
    float v;
    if (d < half) v = (n > 0) ? x[((long)b * N + n - 1) * D + d] : 0.f;
    else          v = x[((long)b * N + n) * D + d];
    vals[i] = v;
    sum += v;
  }
  float mu = block_reduce(sum, red, tid, 128) / (float)D;
  float vs = 0.f;
#pragma unroll
  for (int i = 0; i < 4; ++i) { float t = vals[i] - mu; vs += t * t; }
  float var = block_reduce(vs, red, tid, 128) / (float)D;
  float rstd = rsqrtf(var + 1e-5f);
#pragma unroll
  for (int i = 0; i < 4; ++i) {
    int d = tid + i * 128;
    out[(long)row * D + d] = f2bf((vals[i] - mu) * rstd);
  }
}

// ---------------------------------------------------------------------------
// Kernel 2: fold LN affine into weights; pre-transpose W to [N][K] bf16.
// ---------------------------------------------------------------------------
__global__ void k_fold(const float* __restrict__ W, const float* __restrict__ g,
                       const float* __restrict__ bvec, const float* __restrict__ bias,
                       bf16_t* __restrict__ Wt, float* __restrict__ biasOut,
                       int K, int N) {
  int j = blockIdx.x * blockDim.x + threadIdx.x;
  if (j >= N) return;
  float acc = bias[j];
  for (int k = 0; k < K; ++k) {
    float w = W[(long)k * N + j];
    acc += bvec[k] * w;
    Wt[(long)j * K + k] = f2bf(g[k] * w);
  }
  biasOut[j] = acc;
}

// ---------------------------------------------------------------------------
// Kernel 3: generic bf16 WMMA GEMM with async double-buffered staging:
// C = silu(A[MxK] @ Wt[NxK]^T + bias) -> bf16.  grid(N/128, M/128), block 256.
// Steady-state loop is branch-free; last iteration peeled.
// ---------------------------------------------------------------------------
__global__ void k_gemm_silu(const bf16_t* __restrict__ A, const bf16_t* __restrict__ Bt,
                            const float* __restrict__ bias, bf16_t* __restrict__ C,
                            int M, int N, int K) {
  __shared__ __align__(16) bf16_t sA[2][128 * 32];
  __shared__ __align__(16) bf16_t sB[2][128 * 32];
  int tid = threadIdx.x, wid = tid >> 5, lane = tid & 31;
  int wm = (wid & 3) * 32, wn = (wid >> 2) * 64;
  long bm = (long)blockIdx.y * 128, bn = (long)blockIdx.x * 128;
  const bf16_t* Abase = A + bm * K;
  const bf16_t* Bbase = Bt + bn * K;
  v8f acc[2][4] = {{{}, {}, {}, {}}, {{}, {}, {}, {}}};
  const int T = K >> 5;  // always >= 4 here
  stage_rm_async(sA[0], Abase, K, tid);
  stage_rm_async(sB[0], Bbase, K, tid);
  int t = 0;
  for (; t < T - 1; ++t) {
    stage_rm_async(sA[(t + 1) & 1], Abase + (t + 1) * 32, K, tid);
    stage_rm_async(sB[(t + 1) & 1], Bbase + (t + 1) * 32, K, tid);
    wait_async4();   // tile t's 4 transfers have landed in LDS
    __syncthreads();
    mma_step(sA[t & 1], sB[t & 1], wm, wn, lane, acc);
    __syncthreads();
  }
  wait_async0();
  __syncthreads();
  mma_step(sA[t & 1], sB[t & 1], wm, wn, lane, acc);

  int col0 = lane & 15, rowh = (lane >> 4) * 8;
#pragma unroll
  for (int i = 0; i < 2; ++i)
#pragma unroll
    for (int j = 0; j < 4; ++j) {
      long col = bn + wn + j * 16 + col0;
      float bs = bias[col];
#pragma unroll
      for (int v = 0; v < 8; ++v) {
        long row = bm + wm + i * 16 + rowh + v;
        C[row * N + col] = f2bf(silu_f(acc[i][j][v] + bs));
      }
    }
}

// ---------------------------------------------------------------------------
// Kernel 4: depthwise conv residual: out = s + dwconv1d(s, w), K=17, SAME.
// ---------------------------------------------------------------------------
__global__ void k_dwconv(const bf16_t* __restrict__ s, const float* __restrict__ w,
                         bf16_t* __restrict__ out, int N, int C) {
  long row = blockIdx.x;
  int c = blockIdx.y * blockDim.x + threadIdx.x;
  if (c >= C) return;
  long b = row / N, n = row % N;
  float acc = bf2f(s[row * C + c]);
#pragma unroll
  for (int k = 0; k < 17; ++k) {
    long t = n + k - 8;
    if (t >= 0 && t < N) acc += w[k * C + c] * bf2f(s[(b * N + t) * C + c]);
  }
  out[row * C + c] = f2bf(acc);
}

// ---------------------------------------------------------------------------
// Kernel 5: OffsetScale -> quad_q, lin_q, quad_k, lin_k  (QK=128)
// ---------------------------------------------------------------------------
__global__ void k_scale4(const bf16_t* __restrict__ qk, const float* __restrict__ gamma,
                         const float* __restrict__ beta, bf16_t* __restrict__ qq,
                         bf16_t* __restrict__ lq, bf16_t* __restrict__ kq,
                         bf16_t* __restrict__ lk, long total) {
  long idx = (long)blockIdx.x * blockDim.x + threadIdx.x;
  if (idx >= total) return;
  int d = (int)(idx & 127);
  float q = bf2f(qk[idx]);
  qq[idx] = f2bf(q * gamma[0 * 128 + d] + beta[0 * 128 + d]);
  lq[idx] = f2bf(q * gamma[1 * 128 + d] + beta[1 * 128 + d]);
  kq[idx] = f2bf(q * gamma[2 * 128 + d] + beta[2 * 128 + d]);
  lk[idx] = f2bf(q * gamma[3 * 128 + d] + beta[3 * 128 + d]);
}

// ---------------------------------------------------------------------------
// Kernel 6: per-group quadratic attention matrix: attn = relu(QK^T/G)^2 (bf16)
// grid(2,2,128); K=128; async double-buffered staging, last iteration peeled.
// ---------------------------------------------------------------------------
__global__ void k_quad_attn(const bf16_t* __restrict__ qq, const bf16_t* __restrict__ kq,
                            bf16_t* __restrict__ attn) {
  __shared__ __align__(16) bf16_t sA[2][128 * 32];
  __shared__ __align__(16) bf16_t sB[2][128 * 32];
  const int Kd = 128, G = 256;
  int gg = blockIdx.z;
  int bm = blockIdx.y * 128, bn = blockIdx.x * 128;
  int tid = threadIdx.x, wid = tid >> 5, lane = tid & 31;
  int wm = (wid & 3) * 32, wn = (wid >> 2) * 64;
  const bf16_t* Arow = qq + ((long)gg * G + bm) * Kd;
  const bf16_t* Brow = kq + ((long)gg * G + bn) * Kd;
  v8f acc[2][4] = {{{}, {}, {}, {}}, {{}, {}, {}, {}}};
  const int T = Kd >> 5;  // 4
  stage_rm_async(sA[0], Arow, Kd, tid);
  stage_rm_async(sB[0], Brow, Kd, tid);
  int t = 0;
  for (; t < T - 1; ++t) {
    stage_rm_async(sA[(t + 1) & 1], Arow + (t + 1) * 32, Kd, tid);
    stage_rm_async(sB[(t + 1) & 1], Brow + (t + 1) * 32, Kd, tid);
    wait_async4();
    __syncthreads();
    mma_step(sA[t & 1], sB[t & 1], wm, wn, lane, acc);
    __syncthreads();
  }
  wait_async0();
  __syncthreads();
  mma_step(sA[t & 1], sB[t & 1], wm, wn, lane, acc);

  int col0 = lane & 15, rowh = (lane >> 4) * 8;
#pragma unroll
  for (int i = 0; i < 2; ++i)
#pragma unroll
    for (int j = 0; j < 4; ++j) {
      int col = bn + wn + j * 16 + col0;
#pragma unroll
      for (int v = 0; v < 8; ++v) {
        int row = bm + wm + i * 16 + rowh + v;
        float sv = acc[i][j][v] * (1.f / 256.f);
        sv = sv > 0.f ? sv * sv : 0.f;
        attn[(long)gg * G * G + (long)row * G + col] = f2bf(sv);
      }
    }
}

// ---------------------------------------------------------------------------
// Kernel 7: linear-attention state: kv[b] = (lin_k^T @ hid)/N  (128 x 2048)
// grid(16,1,8); both operands staged with transpose ([k][n] sources).
// ---------------------------------------------------------------------------
__global__ void k_linkv(const bf16_t* __restrict__ lk, const bf16_t* __restrict__ hid,
                        bf16_t* __restrict__ kv) {
  __shared__ __align__(16) bf16_t sA[128 * 32];
  __shared__ __align__(16) bf16_t sB[128 * 32];
  int b = blockIdx.z;
  long bn = (long)blockIdx.x * 128;
  int tid = threadIdx.x, wid = tid >> 5, lane = tid & 31;
  int wm = (wid & 3) * 32, wn = (wid >> 2) * 64;
  v8f acc[2][4] = {{{}, {}, {}, {}}, {{}, {}, {}, {}}};
  for (int kt = 0; kt < 4096; kt += 32) {
    stage_tr(sA, lk + ((long)b * 4096 + kt) * 128, 128, tid);          // A[d][n]
    stage_tr(sB, hid + ((long)b * 4096 + kt) * 2048 + bn, 2048, tid);  // B[e][n]
    __syncthreads();
    mma_step(sA, sB, wm, wn, lane, acc);
    __syncthreads();
  }
  int col0 = lane & 15, rowh = (lane >> 4) * 8;
#pragma unroll
  for (int i = 0; i < 2; ++i)
#pragma unroll
    for (int j = 0; j < 4; ++j) {
      long col = bn + wn + j * 16 + col0;
#pragma unroll
      for (int v = 0; v < 8; ++v) {
        int row = wm + i * 16 + rowh + v;  // d in 0..127
        kv[((long)b * 128 + row) * 2048 + col] = f2bf(acc[i][j][v] * (1.f / 4096.f));
      }
    }
}

// ---------------------------------------------------------------------------
// Kernel 8: att = attn @ hid_group  +  lin_q @ kv[b]    (quad + linear, fused)
// grid(16,2,128); output (BN x 2048) bf16 = [att_v | att_u].
// ---------------------------------------------------------------------------
__global__ void k_quad_apply(const bf16_t* __restrict__ attn, const bf16_t* __restrict__ hid,
                             const bf16_t* __restrict__ lq, const bf16_t* __restrict__ kv,
                             bf16_t* __restrict__ att) {
  __shared__ __align__(16) bf16_t sA[128 * 32];
  __shared__ __align__(16) bf16_t sB[128 * 32];
  int gg = blockIdx.z, b = gg >> 4;
  long bm = (long)blockIdx.y * 128, bn = (long)blockIdx.x * 128;
  int tid = threadIdx.x, wid = tid >> 5, lane = tid & 31;
  int wm = (wid & 3) * 32, wn = (wid >> 2) * 64;
  v8f acc[2][4] = {{{}, {}, {}, {}}, {{}, {}, {}, {}}};
  // quadratic part: K = 256 (keys within group)
  for (int kt = 0; kt < 256; kt += 32) {
    stage_rm(sA, attn + (long)gg * 65536 + bm * 256 + kt, 256, tid);
    stage_tr(sB, hid + ((long)gg * 256 + kt) * 2048 + bn, 2048, tid);
    __syncthreads();
    mma_step(sA, sB, wm, wn, lane, acc);
    __syncthreads();
  }
  // linear part: K = 128 (qk dim)
  for (int kt = 0; kt < 128; kt += 32) {
    stage_rm(sA, lq + ((long)gg * 256 + bm) * 128 + kt, 128, tid);
    stage_tr(sB, kv + ((long)b * 128 + kt) * 2048 + bn, 2048, tid);
    __syncthreads();
    mma_step(sA, sB, wm, wn, lane, acc);
    __syncthreads();
  }
  int col0 = lane & 15, rowh = (lane >> 4) * 8;
#pragma unroll
  for (int i = 0; i < 2; ++i)
#pragma unroll
    for (int j = 0; j < 4; ++j) {
      long col = bn + wn + j * 16 + col0;
#pragma unroll
      for (int v = 0; v < 8; ++v) {
        long row = (long)gg * 256 + bm + wm + i * 16 + rowh + v;
        att[row * 2048 + col] = f2bf(acc[i][j][v]);
      }
    }
}

// ---------------------------------------------------------------------------
// Kernel 9: gating + LayerNorm(1024) (affine folded into W_o).
// ---------------------------------------------------------------------------
__global__ void k_gate_ln(const bf16_t* __restrict__ att, const bf16_t* __restrict__ hid,
                          bf16_t* __restrict__ out) {
  long row = blockIdx.x;
  int tid = threadIdx.x;
  __shared__ float red[256];
  float o[4];
  float sum = 0.f;
#pragma unroll
  for (int i = 0; i < 4; ++i) {
    int e = tid + i * 256;
    float av = bf2f(att[row * 2048 + e]);
    float au = bf2f(att[row * 2048 + 1024 + e]);
    float vv = bf2f(hid[row * 2048 + e]);
    float uu = bf2f(hid[row * 2048 + 1024 + e]);
    float g = au * vv * sigm_f(av * uu);
    o[i] = g;
    sum += g;
  }
  float mu = block_reduce(sum, red, tid, 256) * (1.f / 1024.f);
  float vs = 0.f;
#pragma unroll
  for (int i = 0; i < 4; ++i) { float t = o[i] - mu; vs += t * t; }
  float var = block_reduce(vs, red, tid, 256) * (1.f / 1024.f);
  float rstd = rsqrtf(var + 1e-5f);
#pragma unroll
  for (int i = 0; i < 4; ++i) {
    int e = tid + i * 256;
    out[row * 1024 + e] = f2bf((o[i] - mu) * rstd);
  }
}

// ---------------------------------------------------------------------------
// Kernel 10: final dwconv residual + skip: out = x + s + dwconv(s)  (f32 out)
// ---------------------------------------------------------------------------
__global__ void k_conv_res(const float* __restrict__ x, const bf16_t* __restrict__ s,
                           const float* __restrict__ w, float* __restrict__ out,
                           int N, int C) {
  long row = blockIdx.x;
  int c = blockIdx.y * blockDim.x + threadIdx.x;
  if (c >= C) return;
  long b = row / N, n = row % N;
  float acc = bf2f(s[row * C + c]);
#pragma unroll
  for (int k = 0; k < 17; ++k) {
    long t = n + k - 8;
    if (t >= 0 && t < N) acc += w[k * C + c] * bf2f(s[(b * N + t) * C + c]);
  }
  out[row * C + c] = x[row * C + c] + acc;
}

// ---------------------------------------------------------------------------
// Host launcher
// ---------------------------------------------------------------------------
extern "C" void kernel_launch(void* const* d_in, const int* in_sizes, int n_in,
                              void* d_out, int out_size, void* d_ws, size_t ws_size,
                              hipStream_t stream) {
  (void)in_sizes; (void)n_in; (void)out_size; (void)ws_size;
  const int B = 8, N = 4096, D = 512, H = 2048, QK = 128;
  const long BN = (long)B * N;  // 32768 rows

  const float* x      = (const float*)d_in[0];
  const float* ln_h_g = (const float*)d_in[1];
  const float* ln_h_b = (const float*)d_in[2];
  const float* W_h    = (const float*)d_in[3];
  const float* b_h    = (const float*)d_in[4];
  const float* dw_h   = (const float*)d_in[5];
  const float* ln_q_g = (const float*)d_in[6];
  const float* ln_q_b = (const float*)d_in[7];
  const float* W_qk   = (const float*)d_in[8];
  const float* b_qk   = (const float*)d_in[9];
  const float* dw_qk  = (const float*)d_in[10];
  const float* gamma  = (const float*)d_in[11];
  const float* beta   = (const float*)d_in[12];
  const float* ln_o_g = (const float*)d_in[13];
  const float* ln_o_b = (const float*)d_in[14];
  const float* W_o    = (const float*)d_in[15];
  const float* b_o    = (const float*)d_in[16];
  const float* dw_o   = (const float*)d_in[17];
  float* out = (float*)d_out;

  // --- workspace carve-out (256B aligned) ---
  char* p = (char*)d_ws;
  auto alloc = [&](size_t bytes) {
    void* r = (void*)p;
    p += (bytes + 255) & ~(size_t)255;
    return r;
  };
  bf16_t* A1      = (bf16_t*)alloc(BN * D * 2);            // normalized shifted x
  bf16_t* Wt_h    = (bf16_t*)alloc((size_t)H * D * 2);
  float*  bias_h  = (float*) alloc((size_t)H * 4);
  bf16_t* Wt_qk   = (bf16_t*)alloc((size_t)QK * D * 2);
  float*  bias_qk = (float*) alloc((size_t)QK * 4);
  bf16_t* Wt_o    = (bf16_t*)alloc((size_t)D * (2 * D) * 2);
  float*  bias_o  = (float*) alloc((size_t)D * 4);
  bf16_t* s_h     = (bf16_t*)alloc(BN * H * 2);            // silu(gemm) hidden
  bf16_t* hid     = (bf16_t*)alloc(BN * H * 2);            // + dwconv  -> [v|u]
  bf16_t* s_qk    = (bf16_t*)alloc(BN * QK * 2);
  bf16_t* qkc     = (bf16_t*)alloc(BN * QK * 2);
  bf16_t* qq      = (bf16_t*)alloc(BN * QK * 2);
  bf16_t* lq      = (bf16_t*)alloc(BN * QK * 2);
  bf16_t* kq      = (bf16_t*)alloc(BN * QK * 2);
  bf16_t* lk      = (bf16_t*)alloc(BN * QK * 2);
  bf16_t* attn    = (bf16_t*)alloc((size_t)128 * 256 * 256 * 2);
  bf16_t* kv      = (bf16_t*)alloc((size_t)B * QK * H * 2);
  bf16_t* att     = (bf16_t*)alloc(BN * H * 2);
  bf16_t* gbuf    = (bf16_t*)alloc(BN * (2 * D) * 2);
  bf16_t* s_o     = (bf16_t*)alloc(BN * D * 2);

  // 1. token shift + LN stats
  k_norm_shift<<<dim3((unsigned)BN), 128, 0, stream>>>(x, A1, N, D);

  // 2. fold LN affine into weights (pre-transposed bf16)
  k_fold<<<dim3((H + 127) / 128), 128, 0, stream>>>(W_h, ln_h_g, ln_h_b, b_h, Wt_h, bias_h, D, H);
  k_fold<<<dim3(1), 128, 0, stream>>>(W_qk, ln_q_g, ln_q_b, b_qk, Wt_qk, bias_qk, D, QK);
  k_fold<<<dim3(4), 128, 0, stream>>>(W_o, ln_o_g, ln_o_b, b_o, Wt_o, bias_o, 2 * D, D);

  // 3. hidden = silu(A1 @ W_h' + b') ; then + dwconv
  k_gemm_silu<<<dim3(H / 128, (unsigned)(BN / 128)), 256, 0, stream>>>(A1, Wt_h, bias_h, s_h, (int)BN, H, D);
  k_dwconv<<<dim3((unsigned)BN, H / 256), 256, 0, stream>>>(s_h, dw_h, hid, N, H);

  // 4. qk path
  k_gemm_silu<<<dim3(QK / 128, (unsigned)(BN / 128)), 256, 0, stream>>>(A1, Wt_qk, bias_qk, s_qk, (int)BN, QK, D);
  k_dwconv<<<dim3((unsigned)BN, 1), 256, 0, stream>>>(s_qk, dw_qk, qkc, N, QK);
  {
    long total = BN * QK;
    k_scale4<<<dim3((unsigned)((total + 255) / 256)), 256, 0, stream>>>(qkc, gamma, beta, qq, lq, kq, lk, total);
  }

  // 5. attention
  k_quad_attn<<<dim3(2, 2, 128), 256, 0, stream>>>(qq, kq, attn);
  k_linkv<<<dim3(16, 1, 8), 256, 0, stream>>>(lk, hid, kv);
  k_quad_apply<<<dim3(16, 2, 128), 256, 0, stream>>>(attn, hid, lq, kv, att);

  // 6. gating + output LN
  k_gate_ln<<<dim3((unsigned)BN), 256, 0, stream>>>(att, hid, gbuf);

  // 7. final projection + dwconv + residual
  k_gemm_silu<<<dim3(D / 128, (unsigned)(BN / 128)), 256, 0, stream>>>(gbuf, Wt_o, bias_o, s_o, (int)BN, D, 2 * D);
  k_conv_res<<<dim3((unsigned)BN, D / 256), 256, 0, stream>>>(x, s_o, dw_o, out, N, D);
}
